// NeuralFactorizationMachine_4071628997284
// MI455X (gfx1250) — compile-verified, matrix operand order checked by reading, not run on hardware
//
#include <hip/hip_runtime.h>
#include <hip/hip_bf16.h>

typedef __attribute__((ext_vector_type(16))) _Float16 v16h;
typedef __attribute__((ext_vector_type(8)))  float    v8f;

#define F_FEATS   50000
#define EMBED_DIM 128
#define OUT_DIM   1024
#define BATCH     16
#define TOPK      300

// Monotonic mapping float -> uint32 (order-preserving for all finite floats).
__device__ __forceinline__ unsigned fkey(float v) {
    unsigned u = __float_as_uint(v);
    return (u & 0x80000000u) ? ~u : (u | 0x80000000u);
}

// ---------------------------------------------------------------------------
// Kernel 1: exact top-K per row via 4-pass radix select + 2-phase compaction.
// One block (256 threads) per batch row. LDS histogram + LDS counters.
// ---------------------------------------------------------------------------
__global__ void topk_kernel(const float* __restrict__ X,
                            int* __restrict__ outIdx,
                            float* __restrict__ outVal) {
    const int b = blockIdx.x;
    const float* row = X + (long)b * F_FEATS;
    __shared__ unsigned hist[256];
    __shared__ unsigned s_prefix;
    __shared__ int s_need;
    __shared__ int s_cnt;
    const int t = threadIdx.x;  // 256 threads

    if (t == 0) { s_prefix = 0u; s_need = TOPK; }
    __syncthreads();

    #pragma unroll 1
    for (int pass = 0; pass < 4; ++pass) {
        const int shift = 24 - pass * 8;
        hist[t] = 0u;
        __syncthreads();
        const unsigned prefix = s_prefix;
        for (int f = t; f < F_FEATS; f += 256) {
            unsigned key = fkey(row[f]);
            bool in_class = (pass == 0) || ((key >> (shift + 8)) == prefix);
            if (in_class) atomicAdd(&hist[(key >> shift) & 255u], 1u);
        }
        __syncthreads();
        if (t == 0) {
            int need = s_need;
            int bsel = 0;
            for (int i = 255; i >= 0; --i) {
                int c = (int)hist[i];
                if (need - c <= 0) { bsel = i; break; }
                need -= c;
            }
            s_need = need;                       // still needed inside bucket
            s_prefix = (s_prefix << 8) | (unsigned)bsel;
        }
        __syncthreads();
    }

    const unsigned T = s_prefix;   // exact K-th largest key
    if (t == 0) s_cnt = 0;
    __syncthreads();

    // phase 1: strictly greater than threshold -> always kept
    for (int f = t; f < F_FEATS; f += 256) {
        float v = row[f];
        if (fkey(v) > T) {
            int p = atomicAdd(&s_cnt, 1);
            outIdx[b * TOPK + p] = f;
            outVal[b * TOPK + p] = v;
        }
    }
    __syncthreads();
    // phase 2: equal keys fill the remainder (tie handling)
    for (int f = t; f < F_FEATS; f += 256) {
        float v = row[f];
        if (fkey(v) == T) {
            int p = atomicAdd(&s_cnt, 1);
            if (p < TOPK) { outIdx[b * TOPK + p] = f; outVal[b * TOPK + p] = v; }
        }
    }
}

// ---------------------------------------------------------------------------
// Kernel 2: FM interaction. One block (128 threads) per batch row; thread d
// accumulates sum_emb and sum_sq over the 300 gathered embedding rows
// (fully coalesced: 128 consecutive floats per row). Emits f16 for WMMA.
// ---------------------------------------------------------------------------
__global__ void fm_kernel(const int* __restrict__ idx,
                          const float* __restrict__ val,
                          const float* __restrict__ emb,
                          _Float16* __restrict__ interH) {
    const int b = blockIdx.x;
    const int d = threadIdx.x;  // 128
    float s = 0.f, sq = 0.f;
    #pragma unroll 4
    for (int j = 0; j < TOPK; ++j) {
        float v = val[b * TOPK + j];
        float e = emb[(long)idx[b * TOPK + j] * EMBED_DIM + d];
        s  += v * e;
        sq += (v * v) * (e * e);
    }
    float inter = 0.5f * (s * s - sq);
    interH[b * EMBED_DIM + d] = (_Float16)inter;
}

// ---------------------------------------------------------------------------
// WMMA fragment helpers (layouts per CDNA5 ISA 7.12.2, wave32):
//  A 16x32 f16: lane L -> row m=L&15; halves 0..7 = K[koff..koff+7],
//               halves 8..15 = K[16+koff..], koff = (L>>4)*8.
//  B 32x16 f16: lane L -> col n; same K grouping as A.
//  C/D v8f:     lane L -> col n=L&15; VGPR i -> row m = i + 8*(L>>4).
// ---------------------------------------------------------------------------

// Kernel 3: h = relu(inter @ W1^T + b1); [16,128]x[128,128].
// 1 block, 8 waves, wave w owns N-tile [w*16, w*16+16). K = 128 -> 4 steps.
__global__ void mlp1_wmma(const _Float16* __restrict__ Ah,   // [16][128]
                          const float* __restrict__ W1,      // [128][128]
                          const float* __restrict__ b1,      // [128]
                          _Float16* __restrict__ Hh) {       // [16][128]
    const int wave = threadIdx.x >> 5;
    const int lane = threadIdx.x & 31;
    const int n0   = wave * 16;
    const int m    = lane & 15;
    const int koff = (lane >> 4) * 8;
    const int n    = n0 + m;
    v8f c = {};
    #pragma unroll
    for (int kt = 0; kt < 4; ++kt) {
        const int kb = kt * 32;
        v16h a, bf;
        #pragma unroll
        for (int i = 0; i < 8; ++i) {
            a[i]      = Ah[m * 128 + kb + koff + i];
            a[8 + i]  = Ah[m * 128 + kb + 16 + koff + i];
            bf[i]     = (_Float16)W1[n * 128 + kb + koff + i];
            bf[8 + i] = (_Float16)W1[n * 128 + kb + 16 + koff + i];
        }
        c = __builtin_amdgcn_wmma_f32_16x16x32_f16(false, a, false, bf,
                                                   (short)0, c, false, false);
    }
    const int mbase = (lane >> 4) * 8;
    #pragma unroll
    for (int i = 0; i < 8; ++i) {
        float v = c[i] + b1[n];
        v = v > 0.f ? v : 0.f;
        Hh[(mbase + i) * 128 + n] = (_Float16)v;
    }
}

// Kernel 4: interaction_out = h @ W2^T + b2; [16,128]x[128,1024].
// 8 blocks x 8 waves = 64 waves, one 16x16 N-tile each.
__global__ void mlp2_wmma(const _Float16* __restrict__ Hh,   // [16][128]
                          const float* __restrict__ W2,      // [1024][128]
                          const float* __restrict__ b2,      // [1024]
                          float* __restrict__ outInter) {    // [16][1024]
    const int gw   = blockIdx.x * 8 + (threadIdx.x >> 5);
    const int lane = threadIdx.x & 31;
    const int n0   = gw * 16;
    const int m    = lane & 15;
    const int koff = (lane >> 4) * 8;
    const int n    = n0 + m;
    v8f c = {};
    #pragma unroll
    for (int kt = 0; kt < 4; ++kt) {
        const int kb = kt * 32;
        v16h a, bf;
        #pragma unroll
        for (int i = 0; i < 8; ++i) {
            a[i]      = Hh[m * 128 + kb + koff + i];
            a[8 + i]  = Hh[m * 128 + kb + 16 + koff + i];
            bf[i]     = (_Float16)W2[n * 128 + kb + koff + i];
            bf[8 + i] = (_Float16)W2[n * 128 + kb + 16 + koff + i];
        }
        c = __builtin_amdgcn_wmma_f32_16x16x32_f16(false, a, false, bf,
                                                   (short)0, c, false, false);
    }
    const int mbase = (lane >> 4) * 8;
    #pragma unroll
    for (int i = 0; i < 8; ++i)
        outInter[(mbase + i) * OUT_DIM + n] = c[i] + b2[n];
}

// ---------------------------------------------------------------------------
// Kernel 5: sparse linear: out[b][o] = lin_b[o] + sum_j val*lin_w[o][idx_j].
// One wave per (b,o); lanes split the 300 nnz; butterfly shuffle reduce.
// Prefetch next iteration's gather line (global_prefetch_b8).
// ---------------------------------------------------------------------------
__global__ void linear_kernel(const int* __restrict__ idx,
                              const float* __restrict__ val,
                              const float* __restrict__ lin_w,   // [1024][50000]
                              const float* __restrict__ lin_b,
                              float* __restrict__ outLin) {      // [16][1024]
    const int gw   = blockIdx.x * (blockDim.x >> 5) + (threadIdx.x >> 5);
    const int lane = threadIdx.x & 31;
    const int b    = gw >> 10;          // / OUT_DIM
    const int o    = gw & (OUT_DIM - 1);
    const long roff = (long)o * F_FEATS;
    float acc = 0.f;
    for (int j = lane; j < TOPK; j += 32) {
        int jn = j + 32;
        if (jn < TOPK)
            __builtin_prefetch(&lin_w[roff + idx[b * TOPK + jn]], 0, 0);
        acc += val[b * TOPK + j] * lin_w[roff + idx[b * TOPK + j]];
    }
    #pragma unroll
    for (int mks = 16; mks >= 1; mks >>= 1)
        acc += __shfl_xor(acc, mks, 32);
    if (lane == 0) outLin[b * OUT_DIM + o] = acc + lin_b[o];
}

// Kernel 6: output = linear_out + interaction_out (16384 elements).
__global__ void add_kernel(const float* __restrict__ a,
                           const float* __restrict__ b,
                           float* __restrict__ out, int n) {
    int i = blockIdx.x * blockDim.x + threadIdx.x;
    if (i < n) out[i] = a[i] + b[i];
}

extern "C" void kernel_launch(void* const* d_in, const int* in_sizes, int n_in,
                              void* d_out, int out_size, void* d_ws, size_t ws_size,
                              hipStream_t stream) {
    const float* sae   = (const float*)d_in[0];  // [16][50000]
    const float* emb   = (const float*)d_in[1];  // [50000][128]
    const float* lin_w = (const float*)d_in[2];  // [1024][50000]
    const float* lin_b = (const float*)d_in[3];  // [1024]
    const float* w1    = (const float*)d_in[4];  // [128][128]
    const float* b1    = (const float*)d_in[5];  // [128]
    const float* w2    = (const float*)d_in[6];  // [1024][128]
    const float* b2    = (const float*)d_in[7];  // [1024]

    float* out       = (float*)d_out;            // (output | linear | interaction)
    float* outAll    = out;
    float* outLin    = out + BATCH * OUT_DIM;
    float* outInter  = out + 2 * BATCH * OUT_DIM;

    // workspace carve-up (256B aligned)
    char* ws = (char*)d_ws;
    int*       topIdx = (int*)ws;                                   // 16*300*4
    float*     topVal = (float*)(ws + 19456);                       // 16*300*4
    _Float16*  interH = (_Float16*)(ws + 38912);                    // 16*128*2
    _Float16*  hH     = (_Float16*)(ws + 43264);                    // 16*128*2

    topk_kernel<<<BATCH, 256, 0, stream>>>(sae, topIdx, topVal);
    fm_kernel<<<BATCH, EMBED_DIM, 0, stream>>>(topIdx, topVal, emb, interH);
    mlp1_wmma<<<1, 256, 0, stream>>>(interH, w1, b1, hH);
    mlp2_wmma<<<8, 256, 0, stream>>>(hH, w2, b2, outInter);
    linear_kernel<<<(BATCH * OUT_DIM) / 8, 256, 0, stream>>>(topIdx, topVal,
                                                             lin_w, lin_b, outLin);
    add_kernel<<<(BATCH * OUT_DIM + 255) / 256, 256, 0, stream>>>(outLin, outInter,
                                                                  outAll, BATCH * OUT_DIM);
}
